// FLAREAttention_20426864460085
// MI455X (gfx1250) — matrix operand order repaired and checked by reading, not compiled, not run on hardware
//
#include <hip/hip_runtime.h>
#include <hip/hip_bf16.h>

#define DIM   1024
#define HEADS 16
#define HD    64
#define MLAT  32
#define NSEQ  4096
#define NB    4
#define NTOK  (NB * NSEQ)     // 16384
#define QKV_N (3 * DIM)       // 3072
#define SM_SCALE 0.125f       // HD^-0.5

typedef __attribute__((ext_vector_type(16))) __bf16 v16bf;
typedef __attribute__((ext_vector_type(8)))  __bf16 v8bf;
typedef __attribute__((ext_vector_type(8)))  float  v8f;
typedef __attribute__((ext_vector_type(4)))  unsigned v4u;
typedef __attribute__((ext_vector_type(8)))  int      v8i;
typedef __attribute__((ext_vector_type(4)))  int      v4i;

__device__ __forceinline__ __bf16 bf(float f) { return (__bf16)f; }  // native cvt

__device__ __forceinline__ v8f v8f_zero() {
    v8f z = {0.f, 0.f, 0.f, 0.f, 0.f, 0.f, 0.f, 0.f};
    return z;
}

// Build a 16-element bf16 fragment register from two 16B pieces (ISA WMMA lane layout).
__device__ __forceinline__ v16bf ld16(const __bf16* p0, const __bf16* p1) {
    v8bf lo = *(const v8bf*)p0;
    v8bf hi = *(const v8bf*)p1;
    return __builtin_shufflevector(lo, hi, 0,1,2,3,4,5,6,7,8,9,10,11,12,13,14,15);
}

// A-matrix 16x32 lane layout: lanes 0-15 hold K {0..7,16..23}, lanes 16-31
// hold K {8..15,24..31}. Select by address arithmetic (no divergent branch).
__device__ __forceinline__ v16bf ldfragA(const __bf16* row, int hh) {
    const __bf16* p = row + hh * 8;
    return ld16(p, p + 16);
}
// B-matrix 32x16 lane layout: lanes 0-15 hold K {0..7,16..23}, lanes 16-31
// hold K {8..15,24..31} (same k order as A).
__device__ __forceinline__ v16bf ldfragB(const __bf16* row, int hh) {
    const __bf16* p = row + hh * 8;
    return ld16(p, p + 16);
}

__device__ __forceinline__ v8f wmma_bf16(v16bf a, v16bf b, v8f c) {
    return __builtin_amdgcn_wmma_f32_16x16x32_bf16(false, a, false, b, (short)0, c, false, false);
}

// LDS aperture: flat address low 32 bits ARE the LDS byte address (ISA 10.2).
__device__ __forceinline__ unsigned lds_lo(const void* p) {
    return (unsigned)(unsigned long long)p;
}

#if __has_builtin(__builtin_amdgcn_tensor_load_to_lds)
#define HAVE_TDM 1
// Tensor Data Mover: one instruction DMAs a [tile_h x tile_w] bf16 tile from a
// row-major tensor (row stride in elements) into LDS, packed row-major.
// D# bitfields per CDNA5 ISA sec 8.3/8.4 (2-D: groups 2/3 disabled/zero).
// 6-arg builtin form: (g0, g1, g2, g3, g4, cpol). All words are wave-uniform;
// readfirstlane keeps the descriptor build in SALU.
__device__ __forceinline__ void tdm_load_2d(
        unsigned lds_addr, const void* gaddr, unsigned tile_w, unsigned tile_h,
        unsigned row_stride_elems) {
    unsigned long long ga = (unsigned long long)gaddr;
    unsigned galo = __builtin_amdgcn_readfirstlane((unsigned)(ga & 0xFFFFFFFFu));
    unsigned gahi = __builtin_amdgcn_readfirstlane((unsigned)(ga >> 32));
    unsigned lad  = __builtin_amdgcn_readfirstlane(lds_addr);
    v4u g0;
    g0[0] = 1u;                                          // count=1, user mode
    g0[1] = lad;                                         // lds_addr [63:32]
    g0[2] = galo;                                        // global_addr lo
    g0[3] = (gahi & 0x01FFFFFFu) | (2u << 30);           // addr hi (57b) | type=2
    const unsigned td0 = row_stride_elems;               // tensor_dim0
    const unsigned td1 = 0x00100000u;                    // tensor_dim1 (ample)
    v8i g1;
    g1[0] = (int)(1u << 16);                             // data_size=1 -> 2 bytes
    g1[1] = (int)((td0 & 0xFFFFu) << 16);                // td0[15:0]   @ bits 63:48
    g1[2] = (int)((td0 >> 16) & 0xFFFFu)                 // td0[31:16]  @ bits 79:64
          | (int)((td1 & 0xFFFFu) << 16);                // td1[15:0]   @ bits 95:80
    g1[3] = (int)((td1 >> 16) & 0xFFFFu)                 // td1[31:16]  @ bits 111:96
          | (int)(tile_w << 16);                         // tile_dim0   @ bits 127:112
    g1[4] = (int)(tile_h & 0xFFFFu);                     // tile_dim1 (tile_dim2=0)
    g1[5] = (int)row_stride_elems;                       // dim0_stride[31:0]
    g1[6] = 0;                                           // dim0_stride[47:32], d1s lo
    g1[7] = 0;
    v4i z4 = {0, 0, 0, 0};
    v8i z8 = {0, 0, 0, 0, 0, 0, 0, 0};
    __builtin_amdgcn_tensor_load_to_lds(g0, g1, z4, z4, z8, 0);
}
#else
#define HAVE_TDM 0
// Fallback: CDNA5 async copy, LDS[vdst] = MEM[vaddr], 16B/lane, ASYNCcnt.
__device__ __forceinline__ void async_ld_b128(unsigned lds_off, const void* gptr) {
    asm volatile("global_load_async_to_lds_b128 %0, %1, off"
                 :: "v"(lds_off), "v"(gptr) : "memory");
}
__device__ __forceinline__ void async_ld_b128_o16(unsigned lds_off, const void* gptr) {
    asm volatile("global_load_async_to_lds_b128 %0, %1, off offset:16"
                 :: "v"(lds_off), "v"(gptr) : "memory");
}
#endif

// ---------------------------------------------------------------------------
// One-time converts: x -> bf16; W -> bf16 transposed [col][k] (for B frags).
// ---------------------------------------------------------------------------
__global__ __launch_bounds__(256) void cvt_x_bf16(
        const float* __restrict__ X, __bf16* __restrict__ Xb) {
    int i = (blockIdx.x * 256 + threadIdx.x) * 8;
    float4 f0 = *(const float4*)(X + i);
    float4 f1 = *(const float4*)(X + i + 4);
    v8bf o;
    o[0] = bf(f0.x); o[1] = bf(f0.y); o[2] = bf(f0.z); o[3] = bf(f0.w);
    o[4] = bf(f1.x); o[5] = bf(f1.y); o[6] = bf(f1.z); o[7] = bf(f1.w);
    *(v8bf*)(Xb + i) = o;
}

__global__ __launch_bounds__(256) void cvt_wT_bf16(
        const float* __restrict__ W, __bf16* __restrict__ WT) {
    __shared__ float tile[32][33];
    const int cb = blockIdx.x * 32;            // col block
    const int kb = blockIdx.y * 32;            // k block
    const int tx = threadIdx.x & 31, ty = threadIdx.x >> 5;   // 32 x 8
#pragma unroll
    for (int j = 0; j < 4; ++j)
        tile[ty + 8 * j][tx] = W[(size_t)(kb + ty + 8 * j) * DIM + cb + tx];
    __syncthreads();
#pragma unroll
    for (int j = 0; j < 4; ++j)
        WT[(size_t)(cb + ty + 8 * j) * DIM + kb + tx] = bf(tile[tx][ty + 8 * j]);
}

// ---------------------------------------------------------------------------
// Latent MLP (tiny; batch-independent, computed once).
// ---------------------------------------------------------------------------
__global__ __launch_bounds__(256) void mlp_layer1(
        const float* __restrict__ lq, const float* __restrict__ W1,
        const float* __restrict__ b1, float* __restrict__ h1) {
    int col = blockIdx.x * 256 + threadIdx.x;   // 0..2047
    int row = blockIdx.y;                        // 0..31
    float acc = b1[col];
    const float* a = lq + row * DIM;
    for (int k = 0; k < DIM; ++k)
        acc = fmaf(a[k], W1[k * (2 * DIM) + col], acc);
    acc = 0.5f * acc * (1.0f + erff(acc * 0.70710678118654752f));  // exact GELU
    h1[row * (2 * DIM) + col] = acc;
}

__global__ __launch_bounds__(256) void mlp_layer2(
        const float* __restrict__ h1, const float* __restrict__ W2,
        const float* __restrict__ b2, __bf16* __restrict__ outb) {
    int col = blockIdx.x * 256 + threadIdx.x;   // 0..1023
    int row = blockIdx.y;
    float acc = b2[col];
    const float* a = h1 + row * (2 * DIM);
    for (int k = 0; k < 2 * DIM; ++k)
        acc = fmaf(a[k], W2[k * DIM + col], acc);
    outb[row * DIM + col] = bf(acc);
}

// ---------------------------------------------------------------------------
// Unified bf16 GEMM: C[M x N] = A[M x 1024] @ BT[N x 1024]^T (+bias).
// Block 128x128, 8 waves (4M x 2N), wave tile 32x64. Double-buffered tile
// staging via TDM tensor_load_to_lds (wave 0 issues; TENSORcnt) or per-lane
// global_load_async_to_lds_b128 (ASYNCcnt) fallback. Tail iteration peeled so
// the counter waits are straight-line.
// ---------------------------------------------------------------------------
__global__ __launch_bounds__(256) void gemm_bf16(
        const __bf16* __restrict__ A, const __bf16* __restrict__ BT,
        const float* __restrict__ bias, __bf16* __restrict__ Cb,
        float* __restrict__ Cf, int ldc) {
    __shared__ alignas(16) __bf16 ldsA[2][128 * 32];   // [row][k]
    __shared__ alignas(16) __bf16 ldsB[2][128 * 32];   // [col][k]
    const int tid = threadIdx.x;
    const int lane = tid & 31, wave = tid >> 5;
    const int lanelow = lane & 15;
    const int hh = (lane >= 16) ? 1 : 0;
    const int wm = wave >> 1, wn = wave & 1;
    const int mBase = blockIdx.y * 128;
    const int nBase = blockIdx.x * 128;

    v8f acc[2][4];
#pragma unroll
    for (int i = 0; i < 2; ++i)
#pragma unroll
        for (int j = 0; j < 4; ++j) acc[i][j] = v8f_zero();

#if HAVE_TDM
    auto issue = [&](int buf, int kb) {
        if (wave == 0) {    // TDM ignores EXEC; issue once per workgroup
            tdm_load_2d(lds_lo(&ldsA[buf][0]), A + (size_t)mBase * DIM + kb,
                        32, 128, DIM);
            tdm_load_2d(lds_lo(&ldsB[buf][0]), BT + (size_t)nBase * DIM + kb,
                        32, 128, DIM);
        }
    };
#else
    // per-thread: one row-chunk of A and one of B (32B each, split as 2x16B)
    const int r  = tid >> 1;             // 0..127
    const int kq = (tid & 1) * 16;       // element offset within the 32-wide k
    auto issue = [&](int buf, int kb) {
        const __bf16* ga = A + (size_t)(mBase + r) * DIM + kb + kq;
        const __bf16* gb = BT + (size_t)(nBase + r) * DIM + kb + kq;
        unsigned la = lds_lo(&ldsA[buf][r * 32 + kq]);
        unsigned lb = lds_lo(&ldsB[buf][r * 32 + kq]);
        async_ld_b128(la, ga);
        async_ld_b128_o16(la + 16, ga);
        async_ld_b128(lb, gb);
        async_ld_b128_o16(lb + 16, gb);
    };
#endif

    auto compute = [&](int buf) {
        v16bf afrag[2], bfrag[4];
#pragma unroll
        for (int mt = 0; mt < 2; ++mt)
            afrag[mt] = ldfragA(&ldsA[buf][(wm * 32 + mt * 16 + lanelow) * 32], hh);
#pragma unroll
        for (int nt = 0; nt < 4; ++nt)
            bfrag[nt] = ldfragB(&ldsB[buf][(wn * 64 + nt * 16 + lanelow) * 32], hh);
#pragma unroll
        for (int mt = 0; mt < 2; ++mt)
#pragma unroll
            for (int nt = 0; nt < 4; ++nt)
                acc[mt][nt] = wmma_bf16(afrag[mt], bfrag[nt], acc[mt][nt]);
    };

    const int NK = DIM / 32;             // 32 k-steps (compile-time)
    issue(0, 0);
    for (int ks = 0; ks < NK - 1; ++ks) {
        issue((ks + 1) & 1, (ks + 1) * 32);
#if HAVE_TDM
        if (wave == 0) __builtin_amdgcn_s_wait_tensorcnt(2);  // oldest tile done
#else
        asm volatile("s_wait_asynccnt 4" ::: "memory");       // oldest 4 done
#endif
        __syncthreads();
        compute(ks & 1);
        __syncthreads();
    }
#if HAVE_TDM
    if (wave == 0) __builtin_amdgcn_s_wait_tensorcnt(0);
#else
    asm volatile("s_wait_asynccnt 0" ::: "memory");
#endif
    __syncthreads();
    compute((NK - 1) & 1);

#pragma unroll
    for (int mt = 0; mt < 2; ++mt)
#pragma unroll
        for (int nt = 0; nt < 4; ++nt) {
            int col = nBase + wn * 64 + nt * 16 + lanelow;
            if (Cb) {
#pragma unroll
                for (int v = 0; v < 8; ++v) {
                    int row = mBase + wm * 32 + mt * 16 + v + hh * 8;
                    Cb[(size_t)row * ldc + col] = bf(acc[mt][nt][v]);
                }
            } else {
                float bv = bias[col];
#pragma unroll
                for (int v = 0; v < 8; ++v) {
                    int row = mBase + wm * 32 + mt * 16 + v + hh * 8;
                    Cf[(size_t)row * ldc + col] = acc[mt][nt][v] + bv;
                }
            }
        }
}

// ---------------------------------------------------------------------------
// Encode: latent_rep[b,h] = softmax(Lk @ K^T * s + mask) @ V, flash-streamed.
// One block per (b,h); wave = (stripe 0..3, m-tile 0..1), 32 tokens/iter.
// Output stored TRANSPOSED [d][latent] (bf16) to feed decode B-fragments.
// ---------------------------------------------------------------------------
__global__ __launch_bounds__(256) void encode_attn(
        const __bf16* __restrict__ QKV, const __bf16* __restrict__ LK,
        const unsigned char* __restrict__ mask, __bf16* __restrict__ LrepT) {
    __shared__ alignas(16) __bf16 lds_vT[8][HD * 32];   // per wave: V^T [d][tok]
    __shared__ alignas(16) __bf16 lds_p[8][16 * 32];    // per wave: P [row][tok]
    __shared__ float  lds_O[8][16 * HD];
    __shared__ float  lds_m[8][16];
    __shared__ float  lds_l[8][16];

    const int tid = threadIdx.x, lane = tid & 31, wave = tid >> 5;
    const int lanelow = lane & 15;
    const int hh = (lane >= 16) ? 1 : 0;
    const int bh = blockIdx.x, b = bh / HEADS, h = bh % HEADS;
    const int mt = wave & 1, grp = wave >> 1;
    const int hoff = h * HD;

    v16bf aL[2];   // latent_k A-fragments, K = d split in two 32-steps
#pragma unroll
    for (int kst = 0; kst < 2; ++kst)
        aL[kst] = ldfragA(LK + (mt * 16 + lanelow) * DIM + hoff + kst * 32, hh);

    float m[8], l[8];
    v8f o[4];
#pragma unroll
    for (int v = 0; v < 8; ++v) { m[v] = -INFINITY; l[v] = 0.f; }
#pragma unroll
    for (int nt = 0; nt < 4; ++nt) o[nt] = v8f_zero();

    const int iters = NSEQ / (32 * 4);
    for (int it = 0; it < iters; ++it) {
        const int cb = (it * 4 + grp) * 32;
        {   // stage V^T into per-wave LDS: lane owns one token row
            const __bf16* src = QKV + (size_t)(b * NSEQ + cb + lane) * QKV_N + 2 * DIM + hoff;
#pragma unroll
            for (int g = 0; g < 8; ++g) {
                v8bf t = *(const v8bf*)(src + g * 8);
#pragma unroll
                for (int j = 0; j < 8; ++j)
                    lds_vT[wave][(g * 8 + j) * 32 + lane] = t[j];
            }
        }
        // S = Lk @ K^T  (16 latents x 32 tokens), B = K fed straight from global
        v8f s[2];
        s[0] = v8f_zero(); s[1] = v8f_zero();
#pragma unroll
        for (int kst = 0; kst < 2; ++kst)
#pragma unroll
            for (int nt = 0; nt < 2; ++nt) {
                int tok = b * NSEQ + cb + nt * 16 + lanelow;
                v16bf bk = ldfragB(QKV + (size_t)tok * QKV_N + DIM + hoff + kst * 32, hh);
                s[nt] = wmma_bf16(aL[kst], bk, s[nt]);
            }
#pragma unroll
        for (int nt = 0; nt < 2; ++nt) {
            bool ok = mask[b * NSEQ + cb + nt * 16 + lanelow] != 0;
#pragma unroll
            for (int v = 0; v < 8; ++v)
                s[nt][v] = ok ? s[nt][v] * SM_SCALE : -1e9f;
        }
        // online softmax per row (rows in VGPRs, cols in lanes -> shfl halves)
        float p0[8], p1[8], corr[8];
#pragma unroll
        for (int v = 0; v < 8; ++v) {
            float mx = fmaxf(s[0][v], s[1][v]);
            for (int off = 1; off < 16; off <<= 1) mx = fmaxf(mx, __shfl_xor(mx, off, 32));
            float mn = fmaxf(m[v], mx);
            corr[v] = (m[v] == -INFINITY) ? 0.f : expf(m[v] - mn);
            m[v] = mn;
            p0[v] = expf(s[0][v] - mn);
            p1[v] = expf(s[1][v] - mn);
            float rs = p0[v] + p1[v];
            for (int off = 1; off < 16; off <<= 1) rs += __shfl_xor(rs, off, 32);
            l[v] = l[v] * corr[v] + rs;
        }
#pragma unroll
        for (int nt = 0; nt < 4; ++nt)
#pragma unroll
            for (int v = 0; v < 8; ++v) o[nt][v] *= corr[v];
        // P (C-layout) -> LDS row-major, then reload as A-fragment
#pragma unroll
        for (int v = 0; v < 8; ++v) {
            int r = v + hh * 8;
            lds_p[wave][r * 32 + lanelow]      = bf(p0[v]);
            lds_p[wave][r * 32 + 16 + lanelow] = bf(p1[v]);
        }
        v16bf aP = ldfragA(lds_p[wave] + lanelow * 32, hh);
#pragma unroll
        for (int nt = 0; nt < 4; ++nt) {
            v16bf bv = ldfragB(lds_vT[wave] + (nt * 16 + lanelow) * 32, hh);
            o[nt] = wmma_bf16(aP, bv, o[nt]);
        }
    }
    // dump per-wave partials, merge 4 stripes per m-tile
#pragma unroll
    for (int nt = 0; nt < 4; ++nt)
#pragma unroll
        for (int v = 0; v < 8; ++v)
            lds_O[wave][(v + hh * 8) * HD + nt * 16 + lanelow] = o[nt][v];
    if (lanelow == 0)
#pragma unroll
        for (int v = 0; v < 8; ++v) {
            lds_m[wave][v + hh * 8] = m[v];
            lds_l[wave][v + hh * 8] = l[v];
        }
    __syncthreads();
#pragma unroll
    for (int j = 0; j < 8; ++j) {
        int idx = tid + 256 * j;              // 0..2047 = 32 rows x 64 d
        int row = idx >> 6, d = idx & 63;
        int mtr = row >> 4, lr = row & 15;
        float mg = -INFINITY;
        for (int g = 0; g < 4; ++g) mg = fmaxf(mg, lds_m[2 * g + mtr][lr]);
        float accv = 0.f, lg = 0.f;
        for (int g = 0; g < 4; ++g) {
            int w = 2 * g + mtr;
            float sc = expf(lds_m[w][lr] - mg);
            accv += lds_O[w][lr * HD + d] * sc;
            lg   += lds_l[w][lr] * sc;
        }
        LrepT[((size_t)bh * HD + d) * MLAT + row] = bf(accv / lg);
    }
}

// ---------------------------------------------------------------------------
// Decode: out = softmax(Q @ Lv^T * s) @ latent_rep. Softmax over 32 latents.
// Block = 128 tokens x 1 head; wave = 16 tokens.
// ---------------------------------------------------------------------------
__global__ __launch_bounds__(256) void decode_attn(
        const __bf16* __restrict__ QKV, const __bf16* __restrict__ LV,
        const __bf16* __restrict__ LrepT, __bf16* __restrict__ AttnOut) {
    __shared__ alignas(16) __bf16 lds_p[8][16 * 32];
    const int tid = threadIdx.x, lane = tid & 31, wave = tid >> 5;
    const int lanelow = lane & 15;
    const int hh = (lane >= 16) ? 1 : 0;
    const int h = blockIdx.y;
    const int tokBase = blockIdx.x * 128 + wave * 16;
    const int b = blockIdx.x >> 5;               // 4096/128 = 32 blocks per batch
    const int hoff = h * HD;

    v8f dec[2];
    dec[0] = v8f_zero(); dec[1] = v8f_zero();
#pragma unroll
    for (int kst = 0; kst < 2; ++kst) {
        v16bf aq = ldfragA(QKV + (size_t)(tokBase + lanelow) * QKV_N + hoff + kst * 32, hh);
#pragma unroll
        for (int nt = 0; nt < 2; ++nt) {
            v16bf bv = ldfragB(LV + (nt * 16 + lanelow) * DIM + hoff + kst * 32, hh);
            dec[nt] = wmma_bf16(aq, bv, dec[nt]);
        }
    }
    float sum[8];
#pragma unroll
    for (int v = 0; v < 8; ++v) {
        float a0 = dec[0][v] * SM_SCALE, a1 = dec[1][v] * SM_SCALE;
        float mx = fmaxf(a0, a1);
        for (int off = 1; off < 16; off <<= 1) mx = fmaxf(mx, __shfl_xor(mx, off, 32));
        float e0 = expf(a0 - mx), e1 = expf(a1 - mx);
        float rs = e0 + e1;
        for (int off = 1; off < 16; off <<= 1) rs += __shfl_xor(rs, off, 32);
        sum[v] = rs;
        int r = v + hh * 8;
        lds_p[wave][r * 32 + lanelow]      = bf(e0);
        lds_p[wave][r * 32 + 16 + lanelow] = bf(e1);
    }
    v16bf aP = ldfragA(lds_p[wave] + lanelow * 32, hh);
    v8f outv[4];
#pragma unroll
    for (int nt = 0; nt < 4; ++nt) {
        const __bf16* pb = LrepT + ((size_t)(b * HEADS + h) * HD + nt * 16 + lanelow) * MLAT;
        v16bf blr = ldfragB(pb, hh);            // latent_rep stored transposed
        outv[nt] = wmma_bf16(aP, blr, v8f_zero());
    }
#pragma unroll
    for (int nt = 0; nt < 4; ++nt)
#pragma unroll
        for (int v = 0; v < 8; ++v) {
            int tok = tokBase + v + hh * 8;
            AttnOut[(size_t)tok * DIM + hoff + nt * 16 + lanelow] = bf(outv[nt][v] / sum[v]);
        }
}

// ---------------------------------------------------------------------------
extern "C" void kernel_launch(void* const* d_in, const int* in_sizes, int n_in,
                              void* d_out, int out_size, void* d_ws, size_t ws_size,
                              hipStream_t stream) {
    (void)in_sizes; (void)n_in; (void)out_size; (void)ws_size;
    const float* x            = (const float*)d_in[0];
    const unsigned char* mask = (const unsigned char*)d_in[1];
    const float* lq   = (const float*)d_in[2];
    const float* Wq   = (const float*)d_in[3];
    const float* Wk   = (const float*)d_in[4];
    const float* Wv   = (const float*)d_in[5];
    const float* lkW1 = (const float*)d_in[6];
    const float* lkb1 = (const float*)d_in[7];
    const float* lkW2 = (const float*)d_in[8];
    const float* lkb2 = (const float*)d_in[9];
    const float* lvW1 = (const float*)d_in[10];
    const float* lvb1 = (const float*)d_in[11];
    const float* lvW2 = (const float*)d_in[12];
    const float* lvb2 = (const float*)d_in[13];
    const float* Wout = (const float*)d_in[14];
    const float* bout = (const float*)d_in[15];
    float* out = (float*)d_out;

    char* ws = (char*)d_ws;
    __bf16* qkv   = (__bf16*)ws;  ws += (size_t)NTOK * QKV_N * 2;          // 96 MB
    __bf16* attn  = (__bf16*)ws;  ws += (size_t)NTOK * DIM * 2;            // 32 MB
    __bf16* xb    = (__bf16*)ws;  ws += (size_t)NTOK * DIM * 2;            // 32 MB
    __bf16* wtqkv = (__bf16*)ws;  ws += (size_t)QKV_N * DIM * 2;           // 6 MB
    __bf16* woutT = (__bf16*)ws;  ws += (size_t)DIM * DIM * 2;             // 2 MB
    __bf16* lk    = (__bf16*)ws;  ws += (size_t)MLAT * DIM * 2;
    __bf16* lv    = (__bf16*)ws;  ws += (size_t)MLAT * DIM * 2;
    __bf16* lrepT = (__bf16*)ws;  ws += (size_t)NB * HEADS * HD * MLAT * 2;
    float*  h1    = (float*)ws;   ws += (size_t)MLAT * 2 * DIM * 4;

    dim3 blk(256);
    // one-time bf16 packing (x + transposed weights)
    cvt_x_bf16<<<dim3((NTOK * DIM) / 2048), blk, 0, stream>>>(x, xb);
    cvt_wT_bf16<<<dim3(32, 32), blk, 0, stream>>>(Wq,   wtqkv);
    cvt_wT_bf16<<<dim3(32, 32), blk, 0, stream>>>(Wk,   wtqkv + (size_t)DIM * DIM);
    cvt_wT_bf16<<<dim3(32, 32), blk, 0, stream>>>(Wv,   wtqkv + (size_t)2 * DIM * DIM);
    cvt_wT_bf16<<<dim3(32, 32), blk, 0, stream>>>(Wout, woutT);
    // latent K/V MLPs (batch-independent)
    mlp_layer1<<<dim3((2 * DIM) / 256, MLAT), blk, 0, stream>>>(lq, lkW1, lkb1, h1);
    mlp_layer2<<<dim3(DIM / 256, MLAT),       blk, 0, stream>>>(h1, lkW2, lkb2, lk);
    mlp_layer1<<<dim3((2 * DIM) / 256, MLAT), blk, 0, stream>>>(lq, lvW1, lvb1, h1);
    mlp_layer2<<<dim3(DIM / 256, MLAT),       blk, 0, stream>>>(h1, lvW2, lvb2, lv);
    // fused QKV projection (bf16 WMMA, TDM/async tile staging)
    gemm_bf16<<<dim3(QKV_N / 128, NTOK / 128), blk, 0, stream>>>(
        xb, wtqkv, nullptr, qkv, nullptr, QKV_N);
    // latent encode (flash over n), per (b,h)
    encode_attn<<<dim3(NB * HEADS), blk, 0, stream>>>(qkv, lk, mask, lrepT);
    // token decode over 32 latents
    decode_attn<<<dim3(NTOK / 128, HEADS), blk, 0, stream>>>(qkv, lv, lrepT, attn);
    // output projection + bias
    gemm_bf16<<<dim3(DIM / 128, NTOK / 128), blk, 0, stream>>>(
        attn, woutT, bout, nullptr, out, DIM);
}
